// CrossAttention_188978561658
// MI455X (gfx1250) — matrix operand order repaired and checked by reading, not compiled, not run on hardware
//
#include <hip/hip_runtime.h>
#include <hip/hip_bf16.h>

// ---------------------------------------------------------------------------
// Types for CDNA5 WMMA (wave32): bf16 16x16x32 -> f32
// ---------------------------------------------------------------------------
typedef __bf16 v16bf __attribute__((ext_vector_type(16)));
typedef float  v8f   __attribute__((ext_vector_type(8)));
typedef unsigned int u32x4 __attribute__((ext_vector_type(4)));

union FragV {          // 16 bf16 = 32 bytes = 8 VGPRs (A or B fragment)
    v16bf v;
    u32x4 u[2];
};

__device__ __forceinline__ unsigned short f32_to_bf16(float f) {
    unsigned int u = __float_as_uint(f);
    unsigned int r = u + 0x7FFFu + ((u >> 16) & 1u);   // round-to-nearest-even
    return (unsigned short)(r >> 16);
}

// reductions across the 16-lane half-wave (C/D layout: rows split at lane 16)
__device__ __forceinline__ float half16_max(float x) {
    x = fmaxf(x, __shfl_xor(x, 1, 32));
    x = fmaxf(x, __shfl_xor(x, 2, 32));
    x = fmaxf(x, __shfl_xor(x, 4, 32));
    x = fmaxf(x, __shfl_xor(x, 8, 32));
    return x;
}
__device__ __forceinline__ float half16_sum(float x) {
    x += __shfl_xor(x, 1, 32);
    x += __shfl_xor(x, 2, 32);
    x += __shfl_xor(x, 4, 32);
    x += __shfl_xor(x, 8, 32);
    return x;
}

// ---------------------------------------------------------------------------
// Problem constants
// ---------------------------------------------------------------------------
#define BATCH 2
#define SEQ   2048
#define DMODEL 1024
#define NHEAD 16
#define HDIM  64
#define MROWS (BATCH * SEQ)   // 4096

// ---------------------------------------------------------------------------
// Prep: f32 -> bf16 activation convert
// ---------------------------------------------------------------------------
__global__ void cvt_bf16_kernel(const float* __restrict__ src,
                                unsigned short* __restrict__ dst, int n) {
    int i = blockIdx.x * blockDim.x + threadIdx.x;
    if (i < n) dst[i] = f32_to_bf16(src[i]);
}

// Prep: weight f32 [K][N] -> bf16 transposed Wt[N][K] via LDS tile
__global__ void transpose_bf16_kernel(const float* __restrict__ W,
                                      unsigned short* __restrict__ Wt) {
    __shared__ float tile[32][33];
    int k = blockIdx.y * 32 + threadIdx.y;   // row of W
    int n = blockIdx.x * 32 + threadIdx.x;   // col of W (coalesced read)
    tile[threadIdx.y][threadIdx.x] = W[k * DMODEL + n];
    __syncthreads();
    int nt = blockIdx.x * 32 + threadIdx.y;
    int kt = blockIdx.y * 32 + threadIdx.x;  // coalesced write
    Wt[nt * DMODEL + kt] = f32_to_bf16(tile[threadIdx.x][threadIdx.y]);
}

// ---------------------------------------------------------------------------
// GEMM: C[M=4096][N=1024] = A(bf16,[M][1024]) * W + bias, W given as Wt[n][k]
//   mode 0: bf16 row-major out           (Q, K projections)
//   mode 1: bf16 transposed per batch -> Vt[b][n][s]  (V projection)
//   mode 2: f32 row-major out            (final O projection)
// Each wave: 16x64 tile (4 x 16x16 f32 accumulators), K-step 32.
// ---------------------------------------------------------------------------
__global__ __launch_bounds__(256)
void gemm_bf16_wmma(const unsigned short* __restrict__ A,
                    const unsigned short* __restrict__ Wt,
                    const float* __restrict__ bias,
                    unsigned short* __restrict__ outb,
                    float* __restrict__ outf,
                    int mode) {
    const int K = DMODEL, N = DMODEL;
    const int lane = threadIdx.x & 31;
    const int wave = threadIdx.x >> 5;
    const int hi = lane >> 4;        // half-wave select
    const int ln = lane & 15;
    const int m0 = blockIdx.x * 128 + wave * 16;
    const int n0 = blockIdx.y * 64;

    v8f acc[4] = {};
    const unsigned short* arow = A + (size_t)(m0 + ln) * K;

    for (int k0 = 0; k0 < K; k0 += 32) {
        // A fragment: row (m0+ln), K 0..15 (lo half) / 16..31 (hi half), contiguous
        FragV a;
        a.u[0] = *(const u32x4*)(arow + k0 + hi * 16);
        a.u[1] = *(const u32x4*)(arow + k0 + hi * 16 + 8);
#pragma unroll
        for (int t = 0; t < 4; ++t) {
            // B fragment: column n = n0+t*16+ln == Wt row n, contiguous K slice
            const unsigned short* bp = Wt + (size_t)(n0 + t * 16 + ln) * K + k0 + hi * 16;
            FragV bf;
            bf.u[0] = *(const u32x4*)(bp);
            bf.u[1] = *(const u32x4*)(bp + 8);
            acc[t] = __builtin_amdgcn_wmma_f32_16x16x32_bf16(
                false, a.v, false, bf.v, (short)0, acc[t], false, false);
        }
    }

#pragma unroll
    for (int t = 0; t < 4; ++t) {
        const int col = n0 + t * 16 + ln;
        const float bv = bias[col];
        if (mode == 1) {
            // Vt[b][col][s]: per lane, v=0..7 are 8 consecutive s -> 16B store
            const int b = m0 >> 11;               // m0 / SEQ
            const int sbase = (m0 & (SEQ - 1)) + hi * 8;
            union { u32x4 u; unsigned short s[8]; } pk;
#pragma unroll
            for (int v = 0; v < 8; ++v) pk.s[v] = f32_to_bf16(acc[t][v] + bv);
            *(u32x4*)(outb + ((size_t)(b * N + col)) * SEQ + sbase) = pk.u;
        } else if (mode == 2) {
#pragma unroll
            for (int v = 0; v < 8; ++v)
                outf[(size_t)(m0 + hi * 8 + v) * N + col] = acc[t][v] + bv;
        } else {
#pragma unroll
            for (int v = 0; v < 8; ++v)
                outb[(size_t)(m0 + hi * 8 + v) * N + col] = f32_to_bf16(acc[t][v] + bv);
        }
    }
}

// ---------------------------------------------------------------------------
// Flash attention: per wave one 16-query tile of one (b,h).
//   Qb,Kb: bf16 [b][s][1024] row-major (head slice h*64..h*64+63)
//   Vt:    bf16 [b][d=1024][s=2048] (transposed so PV B-fragments are contiguous)
//   ctx:   bf16 [b][q][1024] merged heads
// K-block = 32 tokens: 4 WMMAs for S, online softmax, 4 WMMAs for P*V.
// ---------------------------------------------------------------------------
__global__ __launch_bounds__(256)
void attention_wmma(const unsigned short* __restrict__ Qb,
                    const unsigned short* __restrict__ Kb,
                    const unsigned short* __restrict__ Vt,
                    unsigned short* __restrict__ ctx) {
    __shared__ unsigned short pbuf[8][16][40];   // per-wave P tile, +8 pad/row

    const int lane = threadIdx.x & 31;
    const int wave = threadIdx.x >> 5;
    const int hi = lane >> 4;
    const int ln = lane & 15;
    const int b = blockIdx.z;
    const int h = blockIdx.y;
    const int q0 = blockIdx.x * 128 + wave * 16;

    // Preload both Q A-fragments (hd = 64 -> two K=32 slices)
    const unsigned short* qrow = Qb + ((size_t)(b * SEQ + q0 + ln)) * DMODEL + h * HDIM;
    FragV qf[2];
#pragma unroll
    for (int s = 0; s < 2; ++s) {
        qf[s].u[0] = *(const u32x4*)(qrow + 32 * s + hi * 16);
        qf[s].u[1] = *(const u32x4*)(qrow + 32 * s + hi * 16 + 8);
    }

    v8f o[4] = {};
    float mrow[8], lrow[8];
#pragma unroll
    for (int v = 0; v < 8; ++v) { mrow[v] = -3.0e38f; lrow[v] = 0.0f; }

    const unsigned short* kbase = Kb + ((size_t)b * SEQ) * DMODEL + h * HDIM;
    const unsigned short* vbase = Vt + ((size_t)(b * DMODEL + h * HDIM)) * SEQ;

    for (int kb0 = 0; kb0 < SEQ; kb0 += 32) {
        // ---- S = Q * K^T, two 16-column tiles
        v8f st[2];
#pragma unroll
        for (int t = 0; t < 2; ++t) {
            v8f sacc = {};
#pragma unroll
            for (int s = 0; s < 2; ++s) {
                // B frag: column n = token kb0+t*16+ln (K rows are hd dims)
                const unsigned short* kp =
                    kbase + (size_t)(kb0 + t * 16 + ln) * DMODEL + 32 * s + hi * 16;
                FragV kf;
                kf.u[0] = *(const u32x4*)(kp);
                kf.u[1] = *(const u32x4*)(kp + 8);
                sacc = __builtin_amdgcn_wmma_f32_16x16x32_bf16(
                    false, qf[s].v, false, kf.v, (short)0, sacc, false, false);
            }
            st[t] = sacc;
        }

        // ---- online softmax per owned row (row = hi*8 + v)
#pragma unroll
        for (int v = 0; v < 8; ++v) {
            float s0 = st[0][v] * 0.125f;            // 1/sqrt(64)
            float s1 = st[1][v] * 0.125f;
            float mx = half16_max(fmaxf(s0, s1));
            float mnew = fmaxf(mrow[v], mx);
            float alpha = __expf(mrow[v] - mnew);
            float p0 = __expf(s0 - mnew);
            float p1 = __expf(s1 - mnew);
            float rs = half16_sum(p0 + p1);
            lrow[v] = lrow[v] * alpha + rs;
            mrow[v] = mnew;
#pragma unroll
            for (int t2 = 0; t2 < 4; ++t2) o[t2][v] *= alpha;
            const int r = hi * 8 + v;
            pbuf[wave][r][ln]      = f32_to_bf16(p0);
            pbuf[wave][r][16 + ln] = f32_to_bf16(p1);
        }

        // ---- reload P (C-layout -> A-layout via LDS), then O += P * V
        FragV pf;
        pf.u[0] = *(const u32x4*)&pbuf[wave][ln][hi * 16];
        pf.u[1] = *(const u32x4*)&pbuf[wave][ln][hi * 16 + 8];
#pragma unroll
        for (int t2 = 0; t2 < 4; ++t2) {
            // B frag: column n = hd dim t2*16+ln, K rows = 32 tokens, contiguous in Vt
            const unsigned short* vp =
                vbase + (size_t)(t2 * 16 + ln) * SEQ + kb0 + hi * 16;
            FragV vf;
            vf.u[0] = *(const u32x4*)(vp);
            vf.u[1] = *(const u32x4*)(vp + 8);
            o[t2] = __builtin_amdgcn_wmma_f32_16x16x32_bf16(
                false, pf.v, false, vf.v, (short)0, o[t2], false, false);
        }
    }

    // ---- normalize and store ctx (merged heads, bf16)
#pragma unroll
    for (int v = 0; v < 8; ++v) {
        const float inv = 1.0f / lrow[v];
        const size_t row = ((size_t)(b * SEQ + q0 + hi * 8 + v)) * DMODEL + h * HDIM;
#pragma unroll
        for (int t2 = 0; t2 < 4; ++t2)
            ctx[row + t2 * 16 + ln] = f32_to_bf16(o[t2][v] * inv);
    }
}

// ---------------------------------------------------------------------------
// Launch
// ---------------------------------------------------------------------------
extern "C" void kernel_launch(void* const* d_in, const int* in_sizes, int n_in,
                              void* d_out, int out_size, void* d_ws, size_t ws_size,
                              hipStream_t stream) {
    const float* Xq  = (const float*)d_in[0];
    const float* Xkv = (const float*)d_in[1];
    const float* Wq  = (const float*)d_in[2];
    const float* bq  = (const float*)d_in[3];
    const float* Wk  = (const float*)d_in[4];
    const float* bk  = (const float*)d_in[5];
    const float* Wv  = (const float*)d_in[6];
    const float* bv  = (const float*)d_in[7];
    const float* Wo  = (const float*)d_in[8];
    const float* bo  = (const float*)d_in[9];
    float* out = (float*)d_out;

    char* ws = (char*)d_ws;
    const size_t MB = 1024 * 1024;
    unsigned short* Xqb = (unsigned short*)(ws + 0 * MB);   // 8 MB
    unsigned short* Xkb = (unsigned short*)(ws + 8 * MB);   // 8 MB
    unsigned short* WqT = (unsigned short*)(ws + 16 * MB);  // 2 MB
    unsigned short* WkT = (unsigned short*)(ws + 18 * MB);  // 2 MB
    unsigned short* WvT = (unsigned short*)(ws + 20 * MB);  // 2 MB
    unsigned short* WoT = (unsigned short*)(ws + 22 * MB);  // 2 MB
    unsigned short* Qb  = (unsigned short*)(ws + 24 * MB);  // 8 MB
    unsigned short* Kb  = (unsigned short*)(ws + 32 * MB);  // 8 MB
    unsigned short* Vt  = (unsigned short*)(ws + 40 * MB);  // 8 MB
    unsigned short* Ctx = (unsigned short*)(ws + 48 * MB);  // 8 MB

    const int nX = BATCH * SEQ * DMODEL;  // 4,194,304
    cvt_bf16_kernel<<<(nX + 255) / 256, 256, 0, stream>>>(Xq,  Xqb, nX);
    cvt_bf16_kernel<<<(nX + 255) / 256, 256, 0, stream>>>(Xkv, Xkb, nX);

    dim3 tb(32, 32), tg(DMODEL / 32, DMODEL / 32);
    transpose_bf16_kernel<<<tg, tb, 0, stream>>>(Wq, WqT);
    transpose_bf16_kernel<<<tg, tb, 0, stream>>>(Wk, WkT);
    transpose_bf16_kernel<<<tg, tb, 0, stream>>>(Wv, WvT);
    transpose_bf16_kernel<<<tg, tb, 0, stream>>>(Wo, WoT);

    dim3 gg(MROWS / 128, DMODEL / 64);  // (32, 16)
    gemm_bf16_wmma<<<gg, 256, 0, stream>>>(Xqb, WqT, bq, Qb, nullptr, 0);
    gemm_bf16_wmma<<<gg, 256, 0, stream>>>(Xkb, WkT, bk, Kb, nullptr, 0);
    gemm_bf16_wmma<<<gg, 256, 0, stream>>>(Xkb, WvT, bv, Vt, nullptr, 1);

    dim3 ag(SEQ / 128, NHEAD, BATCH);   // (16, 16, 2)
    attention_wmma<<<ag, 256, 0, stream>>>(Qb, Kb, Vt, Ctx);

    gemm_bf16_wmma<<<gg, 256, 0, stream>>>(Ctx, WoT, bo, nullptr, out, 2);
}
